// GraphModule_53558242181143
// MI455X (gfx1250) — compile-verified, matrix operand order checked by reading, not run on hardware
//
#include <hip/hip_runtime.h>
#include <stdint.h>

// ---------------------------------------------------------------------------
// EdgeConv x2 for MI455X (gfx1250): bf16 WMMA edge-MLP + f32 atomic scatter.
// v3: software-pipelined B-fragment loads (3-buffer, 2 ahead) + preloaded
//     biases, so WMMAs no longer stall on s_wait_loadcnt 0x0 per K-step.
// ---------------------------------------------------------------------------

typedef __attribute__((ext_vector_type(16))) __bf16 v16bf;
typedef __attribute__((ext_vector_type(8)))  float  v8f;

union BFrag { uint4 q[2]; v16bf v; };

#define WAVES      2        // waves per block (2 x wave32 = 64 threads)
#define ROWS       32       // edges per wave (two 16-row WMMA M-tiles)
#define EB_STRIDE  264      // halves per edge-feature row (256 + 8 pad -> conflict-free b128)
#define HB_STRIDE  136      // halves per hidden row (128 + 8 pad)

__device__ __forceinline__ unsigned short bf16rne(float f) {
  unsigned int u = __float_as_uint(f);
  u += 0x7FFFu + ((u >> 16) & 1u);           // round-to-nearest-even
  return (unsigned short)(u >> 16);
}

__device__ __forceinline__ void load_bfrag(BFrag& B, const unsigned short* base,
                                           int fragIdx, int lane) {
  const unsigned short* wp = base + (size_t)fragIdx * 512 + lane * 16;
  B.q[0] = *(const uint4*)wp;
  B.q[1] = *(const uint4*)(wp + 8);
}

// ---------------------------------------------------------------------------
// Pack W [K x 128] (f32 row-major) into WMMA B-fragment order (bf16).
// Fragment (ks, nt): 32 lanes x 16 halves, contiguous per lane.
// lane: col = nt*16 + (lane&15); half h -> k = ks*32 + ((h&8)<<1) + (lane>>4)*8 + (h&7)
// ---------------------------------------------------------------------------
__global__ void pack_w_kernel(const float* __restrict__ W,
                              unsigned short* __restrict__ out, int K) {
  int t = blockIdx.x * blockDim.x + threadIdx.x;
  int nFrag = (K / 32) * 8;
  if (t >= nFrag * 32) return;
  int lane  = t & 31;
  int frag  = t >> 5;
  int lHalf = lane >> 4;
  int n     = (frag & 7) * 16 + (lane & 15);
  int ks    = frag >> 3;
  unsigned short* dst = out + (size_t)frag * 512 + lane * 16;
#pragma unroll
  for (int h = 0; h < 16; ++h) {
    int k = ks * 32 + ((h & 8) << 1) + lHalf * 8 + (h & 7);
    dst[h] = bf16rne(W[(size_t)k * 128 + n]);
  }
}

__global__ void zero_f32_kernel(float* __restrict__ p, long long n) {
  long long i = blockIdx.x * (long long)blockDim.x + threadIdx.x;
  long long s = gridDim.x * (long long)blockDim.x;
  for (; i < n; i += s) p[i] = 0.0f;
}

// ---------------------------------------------------------------------------
// One wave handles 32 edges (2 M-tiles): gather -> [x_i | x_j-x_i] (bf16,
// LDS) -> GEMM1 (32x256 @ 256x128, 128 WMMA) -> ReLU -> GEMM2 (32x128 @
// 128x128, 64 WMMA) -> nontemporal msg store + f32 atomic scatter-add.
// B fragments are loaded once per (nt,ks), consumed by both M-tiles, and
// software-pipelined two iterations ahead.
// ---------------------------------------------------------------------------
template <bool WRITE_MSG>
__global__ __launch_bounds__(WAVES * 32, 1)
void edge_conv_kernel(const float* __restrict__ x,
                      const long long* __restrict__ ei,   // [2, E] int64
                      const unsigned short* __restrict__ pW1,
                      const float* __restrict__ b1,
                      const unsigned short* __restrict__ pW2,
                      const float* __restrict__ b2,
                      float* __restrict__ featOut,
                      float* __restrict__ msgOut,
                      int E) {
  __shared__ unsigned short edgebuf[WAVES][ROWS * EB_STRIDE];  // 33792 B
  __shared__ unsigned short hidbuf [WAVES][ROWS * HB_STRIDE];  // 17408 B
  __shared__ int            idxS   [WAVES][2 * ROWS];          // src[0..31], dst[32..63]

  const int lane  = threadIdx.x & 31;
  const int wave  = threadIdx.x >> 5;
  const int lHalf = lane >> 4;
  const int colN  = lane & 15;

  const int numTiles = E / ROWS;
  int tile = blockIdx.x * WAVES + wave;
  if (tile >= numTiles) tile = numTiles - 1;   // E % (ROWS*WAVES) == 0 here
  const int e0 = tile * ROWS;

  // Stage the 32 edge indices (int64 -> int).
  idxS[wave][lane]        = (int)ei[e0 + lane];                // src
  idxS[wave][ROWS + lane] = (int)ei[(size_t)E + e0 + lane];    // dst
  __syncthreads();

  // Preload per-lane biases for all 8 N-tiles of both GEMMs.
  float bias1[8], bias2[8];
#pragma unroll
  for (int nt = 0; nt < 8; ++nt) {
    bias1[nt] = b1[nt * 16 + colN];
    bias2[nt] = b2[nt * 16 + colN];
  }

  // Gather rows and build edge features [x_i | x_j - x_i] in bf16.
  // 32 lanes cover one 128-f32 row as float4 each.
  for (int r = 0; r < ROWS; ++r) {
    int di = idxS[wave][ROWS + r];
    int si = idxS[wave][r];
    const float4 a = ((const float4*)(x + (size_t)di * 128))[lane];  // x_i
    const float4 b = ((const float4*)(x + (size_t)si * 128))[lane];  // x_j
    unsigned int u0 = (unsigned int)bf16rne(a.x) | ((unsigned int)bf16rne(a.y) << 16);
    unsigned int u1 = (unsigned int)bf16rne(a.z) | ((unsigned int)bf16rne(a.w) << 16);
    unsigned int d0 = (unsigned int)bf16rne(b.x - a.x) | ((unsigned int)bf16rne(b.y - a.y) << 16);
    unsigned int d1 = (unsigned int)bf16rne(b.z - a.z) | ((unsigned int)bf16rne(b.w - a.w) << 16);
    *(uint2*)&edgebuf[wave][r * EB_STRIDE + lane * 4]       = uint2{u0, u1};
    *(uint2*)&edgebuf[wave][r * EB_STRIDE + 128 + lane * 4] = uint2{d0, d1};
  }
  __syncthreads();

  // A fragments for GEMM1: 2 M-tiles x 8 K-steps, cached in VGPRs (128 regs).
  BFrag A[2][8];
#pragma unroll
  for (int mt = 0; mt < 2; ++mt) {
#pragma unroll
    for (int ks = 0; ks < 8; ++ks) {
      const unsigned short* p =
          &edgebuf[wave][(mt * 16 + colN) * EB_STRIDE + ks * 32 + lHalf * 8];
      A[mt][ks].q[0] = *(const uint4*)p;
      A[mt][ks].q[1] = *(const uint4*)(p + 16);
    }
  }

  // GEMM1 + bias + ReLU -> hidden (bf16 LDS).
  // Flattened (nt,ks) pipeline: B fragment f consumed while f+1, f+2 load.
  {
    BFrag B[3];
    load_bfrag(B[0], pW1, /*ks=*/0 * 8 + /*nt=*/0, lane);   // f = 0
    load_bfrag(B[1], pW1, /*ks=*/1 * 8 + /*nt=*/0, lane);   // f = 1
#pragma unroll
    for (int nt = 0; nt < 8; ++nt) {
      float bv = bias1[nt];
      v8f acc0 = {bv, bv, bv, bv, bv, bv, bv, bv};
      v8f acc1 = acc0;
#pragma unroll
      for (int ks = 0; ks < 8; ++ks) {
        const int f = nt * 8 + ks;
        if (f + 2 < 64) {
          const int f2 = f + 2;
          const int nt2 = f2 >> 3, ks2 = f2 & 7;
          load_bfrag(B[(f + 2) % 3], pW1, ks2 * 8 + nt2, lane);
        }
        const BFrag& Bc = B[f % 3];
        acc0 = __builtin_amdgcn_wmma_f32_16x16x32_bf16(
            false, A[0][ks].v, false, Bc.v, (short)0, acc0, false, false);
        acc1 = __builtin_amdgcn_wmma_f32_16x16x32_bf16(
            false, A[1][ks].v, false, Bc.v, (short)0, acc1, false, false);
      }
#pragma unroll
      for (int v = 0; v < 8; ++v) {
        float r0 = fmaxf(acc0[v], 0.0f);
        float r1 = fmaxf(acc1[v], 0.0f);
        hidbuf[wave][(v + 8 * lHalf) * HB_STRIDE + nt * 16 + colN]      = bf16rne(r0);
        hidbuf[wave][(16 + v + 8 * lHalf) * HB_STRIDE + nt * 16 + colN] = bf16rne(r1);
      }
    }
  }
  __syncthreads();

  // A fragments for GEMM2 (hidden 32x128 bf16): 2 M-tiles x 4 K-steps.
  BFrag H[2][4];
#pragma unroll
  for (int mt = 0; mt < 2; ++mt) {
#pragma unroll
    for (int ks = 0; ks < 4; ++ks) {
      const unsigned short* p =
          &hidbuf[wave][(mt * 16 + colN) * HB_STRIDE + ks * 32 + lHalf * 8];
      H[mt][ks].q[0] = *(const uint4*)p;
      H[mt][ks].q[1] = *(const uint4*)(p + 16);
    }
  }

  // This lane's destination node ids (C layout rows m = mt*16 + 8*lHalf + v).
  int dsts[2][8];
#pragma unroll
  for (int mt = 0; mt < 2; ++mt)
#pragma unroll
    for (int v = 0; v < 8; ++v)
      dsts[mt][v] = idxS[wave][ROWS + mt * 16 + 8 * lHalf + v];

  // GEMM2 + bias -> msg; scatter-add into featOut. Same B pipeline (32 frags).
  {
    BFrag B[3];
    load_bfrag(B[0], pW2, /*ks=*/0 * 8 + /*nt=*/0, lane);   // f = 0
    load_bfrag(B[1], pW2, /*ks=*/1 * 8 + /*nt=*/0, lane);   // f = 1
#pragma unroll
    for (int nt = 0; nt < 8; ++nt) {
      float bv = bias2[nt];
      v8f acc0 = {bv, bv, bv, bv, bv, bv, bv, bv};
      v8f acc1 = acc0;
#pragma unroll
      for (int ks = 0; ks < 4; ++ks) {
        const int f = nt * 4 + ks;
        if (f + 2 < 32) {
          const int f2 = f + 2;
          const int nt2 = f2 >> 2, ks2 = f2 & 3;
          load_bfrag(B[(f + 2) % 3], pW2, ks2 * 8 + nt2, lane);
        }
        const BFrag& Bc = B[f % 3];
        acc0 = __builtin_amdgcn_wmma_f32_16x16x32_bf16(
            false, H[0][ks].v, false, Bc.v, (short)0, acc0, false, false);
        acc1 = __builtin_amdgcn_wmma_f32_16x16x32_bf16(
            false, H[1][ks].v, false, Bc.v, (short)0, acc1, false, false);
      }
      const int n = nt * 16 + colN;
#pragma unroll
      for (int mt = 0; mt < 2; ++mt) {
#pragma unroll
        for (int v = 0; v < 8; ++v) {
          const int m = mt * 16 + v + 8 * lHalf;
          const float val = (mt == 0) ? acc0[v] : acc1[v];
          if (WRITE_MSG)   // streaming output: keep out of L2 (x/feat stay hot)
            __builtin_nontemporal_store(val, msgOut + (size_t)(e0 + m) * 128 + n);
          atomicAdd(featOut + (size_t)dsts[mt][v] * 128 + n, val);
        }
      }
    }
  }
}

// ---------------------------------------------------------------------------
// Host launcher.
// ---------------------------------------------------------------------------
extern "C" void kernel_launch(void* const* d_in, const int* in_sizes, int n_in,
                              void* d_out, int out_size, void* d_ws, size_t ws_size,
                              hipStream_t stream) {
  const float*     x    = (const float*)d_in[0];
  const long long* ei   = (const long long*)d_in[1];     // int64 [2,E]
  const float*     W1_0 = (const float*)d_in[2];
  const float*     b1_0 = (const float*)d_in[3];
  const float*     W2_0 = (const float*)d_in[4];
  const float*     b2_0 = (const float*)d_in[5];
  const float*     W1_1 = (const float*)d_in[6];
  const float*     b1_1 = (const float*)d_in[7];
  const float*     W2_1 = (const float*)d_in[8];
  const float*     b2_1 = (const float*)d_in[9];

  const int D = 128;
  const int N = in_sizes[0] / D;      // 50000
  const int E = in_sizes[1] / 2;      // 800000

  // Workspace layout: packed bf16 weights, then feat1 (layer-1 node output).
  char* ws = (char*)d_ws;
  unsigned short* pW1_0 = (unsigned short*)(ws);                       // 64 KB
  unsigned short* pW2_0 = (unsigned short*)(ws + 65536);               // 32 KB
  unsigned short* pW1_1 = (unsigned short*)(ws + 65536 + 32768);       // 64 KB
  unsigned short* pW2_1 = (unsigned short*)(ws + 2 * 65536 + 32768);   // 32 KB
  float*          feat1 = (float*)(ws + 2 * 65536 + 2 * 32768);        // N*128 f32

  // Pre-pack weights into WMMA B-fragment order (tiny, one-shot).
  pack_w_kernel<<<(2048 + 127) / 128, 128, 0, stream>>>(W1_0, pW1_0, 256);
  pack_w_kernel<<<(1024 + 127) / 128, 128, 0, stream>>>(W2_0, pW2_0, 128);
  pack_w_kernel<<<(2048 + 127) / 128, 128, 0, stream>>>(W1_1, pW1_1, 256);
  pack_w_kernel<<<(1024 + 127) / 128, 128, 0, stream>>>(W2_1, pW2_1, 128);

  // Zero accumulation targets (feat1 in ws, final feat in d_out).
  zero_f32_kernel<<<768, 256, 0, stream>>>(feat1, (long long)N * D);
  zero_f32_kernel<<<768, 256, 0, stream>>>((float*)d_out, (long long)N * D);

  const int blocks = (E / ROWS + WAVES - 1) / WAVES;   // 12500
  // Layer 0: x -> feat1 (no msg output needed).
  edge_conv_kernel<false><<<blocks, WAVES * 32, 0, stream>>>(
      x, ei, pW1_0, b1_0, pW2_0, b2_0, feat1, nullptr, E);
  // Layer 1: feat1 -> d_out feat region; msg -> d_out msg region.
  edge_conv_kernel<true><<<blocks, WAVES * 32, 0, stream>>>(
      feat1, ei, pW1_1, b1_1, pW2_1, b2_1,
      (float*)d_out, (float*)d_out + (size_t)N * D, E);
}
